// AttentionModule_74612171866182
// MI455X (gfx1250) — matrix operand order repaired and checked by reading, not compiled, not run on hardware
//
#include <hip/hip_runtime.h>
#include <stdint.h>

// ---------------------------------------------------------------------------
// Causal MHA for MI455X (gfx1250, wave32, WMMA + async global->LDS staging).
//   out = proj( softmax_causal( Q K^T / sqrt(d) ) V ),  QKV = x @ w_qkv + b
// Pipeline is f16-resident: one conversion pass, then every tile staged into
// LDS with GLOBAL_LOAD_ASYNC_TO_LDS_B128 (ASYNCcnt) and consumed by
// v_wmma_f32_16x16x32_f16.  Accumulation/softmax stay in f32.
// ---------------------------------------------------------------------------

typedef __attribute__((ext_vector_type(16))) _Float16 v16h;
typedef __attribute__((ext_vector_type(4)))  _Float16 v4h;
typedef __attribute__((ext_vector_type(2)))  _Float16 h2;
typedef __attribute__((ext_vector_type(8)))  float    v8f;
typedef __attribute__((ext_vector_type(4)))  int      v4i;

#define D_MODEL 1024
#define N_HEADS 16
#define HEAD_DIM 64
#define T_SEQ 2048
#define C3 (3 * D_MODEL)
#define NEGBIG (-1.0e30f)

#if __has_builtin(__builtin_amdgcn_global_load_async_to_lds_b128)
#define HAVE_ASYNC_LDS 1
#else
#define HAVE_ASYNC_LDS 0
#endif

// 16-byte global -> LDS copy. Async (ASYNCcnt-tracked DMA) when available.
static __device__ __forceinline__ void async_cp16(_Float16* lds_dst,
                                                  const _Float16* gsrc) {
#if HAVE_ASYNC_LDS
  __builtin_amdgcn_global_load_async_to_lds_b128(
      (__attribute__((address_space(1))) v4i*)(uintptr_t)gsrc,
      (__attribute__((address_space(3))) v4i*)(uint32_t)(uintptr_t)lds_dst,
      0, 0);
#else
  *(v4i*)lds_dst = *(const v4i*)gsrc;
#endif
}

static __device__ __forceinline__ void wait_async_0() {
#if HAVE_ASYNC_LDS
#if __has_builtin(__builtin_amdgcn_s_wait_asynccnt)
  __builtin_amdgcn_s_wait_asynccnt(0);
#else
  asm volatile("s_wait_asynccnt 0x0" ::: "memory");
#endif
#endif
}

static __device__ __forceinline__ void wait_async_3() {
#if HAVE_ASYNC_LDS
#if __has_builtin(__builtin_amdgcn_s_wait_asynccnt)
  __builtin_amdgcn_s_wait_asynccnt(3);
#else
  asm volatile("s_wait_asynccnt 0x3" ::: "memory");
#endif
#endif
}

static __device__ __forceinline__ v8f wmma16x16x32(v16h a, v16h b, v8f c) {
  return __builtin_amdgcn_wmma_f32_16x16x32_f16(
      /*neg_a=*/false, a, /*neg_b=*/false, b,
      /*c_mod=*/(short)0, c, /*reuse_a=*/false, /*reuse_b=*/false);
}

// A-fragment (16x32 f16): lane = 16*g + m holds row m; VGPR i holds K pair
// starting at kA(i,g).  (ISA 7.12.2 16-bit A layout)
static __device__ __forceinline__ int kA(int i, int g) {
  return ((i & 3) * 2) + ((i >> 2) * 16) + 8 * g;
}
// B-fragment (32x16 f16): lane = 16*g + n holds col n; VGPR i holds K pair
// starting at kB(i,g).
static __device__ __forceinline__ int kB(int i, int g) {
  return ((i & 3) * 2) + ((i >> 2) * 16) + 8 * g;
}

static __device__ __forceinline__ v16h load_afrag(const _Float16* base, int ld,
                                                  int lane, int kofs) {
  int m = lane & 15, g = lane >> 4;
  v16h a;
#pragma unroll
  for (int i = 0; i < 8; ++i) {
    h2 p = *(const h2*)&base[m * ld + kofs + kA(i, g)];
    a[2 * i] = p[0];
    a[2 * i + 1] = p[1];
  }
  return a;
}

static __device__ __forceinline__ v16h load_bfrag(const _Float16* base, int ld,
                                                  int lane, int kofs) {
  int n = lane & 15, g = lane >> 4;
  v16h b;
#pragma unroll
  for (int i = 0; i < 8; ++i) {
    h2 p = *(const h2*)&base[n * ld + kofs + kB(i, g)];
    b[2 * i] = p[0];
    b[2 * i + 1] = p[1];
  }
  return b;
}

static __device__ __forceinline__ v8f zero_v8f() {
  v8f z = {0.f, 0.f, 0.f, 0.f, 0.f, 0.f, 0.f, 0.f};
  return z;
}

// ---------------------------------------------------------------------------
// f32 -> f16 elementwise conversion (x).
// ---------------------------------------------------------------------------
__global__ __launch_bounds__(256) void cvt_f32_to_f16(
    const float* __restrict__ in, _Float16* __restrict__ out, int n4) {
  int i = blockIdx.x * 256 + threadIdx.x;
  if (i < n4) {
    float4 v = *(const float4*)(in + (size_t)i * 4);
    v4h o = {(_Float16)v.x, (_Float16)v.y, (_Float16)v.z, (_Float16)v.w};
    *(v4h*)(out + (size_t)i * 4) = o;
  }
}

// ---------------------------------------------------------------------------
// f32 [K][N] -> f16 [N][K] transpose+convert (weights), 64x64 LDS tiles.
// ---------------------------------------------------------------------------
__global__ __launch_bounds__(256) void transpose_cvt_f16(
    const float* __restrict__ in, _Float16* __restrict__ out, int K, int N) {
  __shared__ _Float16 tile[64][65];
  const int k0 = blockIdx.y * 64, n0 = blockIdx.x * 64;
  const int tid = threadIdx.x;
#pragma unroll
  for (int r = 0; r < 16; ++r) {
    int idx = tid + 256 * r;
    int kr = idx >> 6, nc = idx & 63;
    tile[kr][nc] = (_Float16)in[(size_t)(k0 + kr) * N + n0 + nc];
  }
  __syncthreads();
#pragma unroll
  for (int r = 0; r < 16; ++r) {
    int idx = tid + 256 * r;
    int nr = idx >> 6, kc = idx & 63;
    out[(size_t)(n0 + nr) * K + k0 + kc] = tile[kc][nr];
  }
}

// ---------------------------------------------------------------------------
// GEMM:  C[M,N] = A[M,K] @ Bt[N,K]^T + bias[N]   (f16 inputs, f32 accum)
// Block = 256 threads (8 waves), tile 128x64, double-buffered async staging.
// EPI 0: f32 output [M][N].
// EPI 1: qkv epilogue: cols < 2C -> f16 qkv_h[M][3C]; cols >= 2C (the V part)
//        scattered transposed into vT[(b*H+h)*64+d][T] (f16) for attention.
// ---------------------------------------------------------------------------
template <int EPI>
__global__ __launch_bounds__(256) void gemm_f16(
    const _Float16* __restrict__ A, const _Float16* __restrict__ Bt,
    const float* __restrict__ bias, void* __restrict__ Cout,
    _Float16* __restrict__ vT, int M, int N, int K) {
  __shared__ _Float16 Asm[2][128 * 32];  // [m][k]
  __shared__ _Float16 Bsm[2][64 * 32];   // [n][k]

  const int tid = threadIdx.x;
  const int lane = tid & 31;
  const int w = tid >> 5;
  const int wm = (w & 3) * 32;
  const int wn = (w >> 2) * 32;
  const int m0 = blockIdx.y * 128;
  const int n0 = blockIdx.x * 64;

  // Stage one 128x32 A tile + 64x32 B tile: 3 x 16B async ops per thread.
  auto stage = [&](int buf, int k0) {
#pragma unroll
    for (int r = 0; r < 2; ++r) {
      int idx = tid + 256 * r;  // 512 A chunks: 128 rows x 4 x 16B
      int row = idx >> 2;
      int cf = (idx & 3) << 3;
      async_cp16(&Asm[buf][row * 32 + cf],
                 A + (size_t)(m0 + row) * K + k0 + cf);
    }
    {
      int n = tid >> 2;         // 256 B chunks: 64 rows x 4 x 16B
      int cf = (tid & 3) << 3;
      async_cp16(&Bsm[buf][n * 32 + cf],
                 Bt + (size_t)(n0 + n) * K + k0 + cf);
    }
  };

  v8f acc[2][2];
#pragma unroll
  for (int a = 0; a < 2; ++a)
#pragma unroll
    for (int bI = 0; bI < 2; ++bI) acc[a][bI] = zero_v8f();

  const int ntiles = K >> 5;
  stage(0, 0);
  for (int kt = 0; kt < ntiles; ++kt) {
    const int cur = kt & 1;
    if (kt + 1 < ntiles) {
      stage(cur ^ 1, (kt + 1) * 32);  // prefetch next tile (other buffer)
      wait_async_3();                 // retire current tile's 3 copies
    } else {
      wait_async_0();
    }
    __syncthreads();  // everyone's copies for tile kt landed

    v16h a0 = load_afrag(&Asm[cur][(wm + 0) * 32], 32, lane, 0);
    v16h a1 = load_afrag(&Asm[cur][(wm + 16) * 32], 32, lane, 0);
    v16h b0 = load_bfrag(&Bsm[cur][(wn + 0) * 32], 32, lane, 0);
    v16h b1 = load_bfrag(&Bsm[cur][(wn + 16) * 32], 32, lane, 0);
    acc[0][0] = wmma16x16x32(a0, b0, acc[0][0]);
    acc[0][1] = wmma16x16x32(a0, b1, acc[0][1]);
    acc[1][0] = wmma16x16x32(a1, b0, acc[1][0]);
    acc[1][1] = wmma16x16x32(a1, b1, acc[1][1]);

    __syncthreads();  // done reading tile kt before its buffer is re-staged
  }

  // Epilogue (C layout: lane 16*g+n holds col n, VGPR i holds row i+8*g).
  const int nl = lane & 15, g = lane >> 4;
#pragma unroll
  for (int mf = 0; mf < 2; ++mf)
#pragma unroll
    for (int nf = 0; nf < 2; ++nf)
#pragma unroll
      for (int i = 0; i < 8; ++i) {
        int row = m0 + wm + mf * 16 + i + 8 * g;
        int col = n0 + wn + nf * 16 + nl;
        float v = acc[mf][nf][i] + bias[col];
        if (EPI == 0) {
          ((float*)Cout)[(size_t)row * N + col] = v;
        } else {
          if (col < 2 * D_MODEL) {  // uniform per block (n0 % 64 == 0)
            ((_Float16*)Cout)[(size_t)row * C3 + col] = (_Float16)v;
          } else {
            int c2 = col - 2 * D_MODEL;
            int hh = c2 >> 6, dd = c2 & 63;
            int bb = row >> 11, tt = row & 2047;
            vT[(size_t)((bb * N_HEADS + hh) * HEAD_DIM + dd) * T_SEQ + tt] =
                (_Float16)v;
          }
        }
      }
}

// ---------------------------------------------------------------------------
// Flash attention, causal.  One block (128 thr / 4 waves) per (b, h, 64-row
// query tile).  K tile staged async from qkv_h [tok][3C]; V tile staged async
// from pre-transposed vT [(b*H+h)*64+d][T].  Output written f16.
// ---------------------------------------------------------------------------
__global__ __launch_bounds__(128) void attn_causal(
    const _Float16* __restrict__ qkvh, const _Float16* __restrict__ vT,
    _Float16* __restrict__ attn_out) {
  __shared__ _Float16 Ksm[64 * 64];     // [key][d]  -> B-frag layout for QK^T
  __shared__ _Float16 Vsm[64 * 64];     // [d][key]  -> B-frag layout for P.V
  __shared__ _Float16 Psm[4][16 * 64];  // per-wave P tile [q][key]

  const int tid = threadIdx.x;
  const int lane = tid & 31;
  const int w = tid >> 5;
  const int nl = lane & 15, g = lane >> 4;
  const int qt = blockIdx.x;
  const int h = blockIdx.y;
  const int b = blockIdx.z;
  const int bT = b * T_SEQ;
  const size_t vbase = (size_t)(b * N_HEADS + h) * HEAD_DIM;

  // Q fragments (f16 direct from global), kept in registers.
  v16h qf[2];
  {
    const int qrow = qt * 64 + w * 16 + nl;
    const _Float16* qp = qkvh + (size_t)(bT + qrow) * C3 + h * HEAD_DIM;
#pragma unroll
    for (int ks = 0; ks < 2; ++ks)
#pragma unroll
      for (int i = 0; i < 8; ++i) {
        h2 p = *(const h2*)&qp[ks * 32 + kA(i, g)];
        qf[ks][2 * i] = p[0];
        qf[ks][2 * i + 1] = p[1];
      }
  }

  v8f oacc[4];
#pragma unroll
  for (int t = 0; t < 4; ++t) oacc[t] = zero_v8f();
  float mrow[8], lrow[8];
#pragma unroll
  for (int i = 0; i < 8; ++i) { mrow[i] = NEGBIG; lrow[i] = 0.f; }

  const float scale = 0.125f;  // 1/sqrt(64)

  for (int kb = 0; kb <= qt; ++kb) {
    __syncthreads();  // previous iteration done reading Ksm/Vsm
    // Stage K tile (64 keys x 128B) and V tile (64 d x 128B): async DMA.
#pragma unroll
    for (int r = 0; r < 4; ++r) {
      int idx = tid + 128 * r;
      int key = idx >> 3;
      int cf = (idx & 7) << 3;
      async_cp16(&Ksm[key * 64 + cf],
                 qkvh + (size_t)(bT + kb * 64 + key) * C3 + D_MODEL +
                     h * HEAD_DIM + cf);
    }
#pragma unroll
    for (int r = 0; r < 4; ++r) {
      int idx = tid + 128 * r;
      int d = idx >> 3;
      int cf = (idx & 7) << 3;
      async_cp16(&Vsm[d * 64 + cf],
                 vT + (vbase + d) * T_SEQ + kb * 64 + cf);
    }
    wait_async_0();
    __syncthreads();

    // --- S = Q K^T.
    v8f s[4];
#pragma unroll
    for (int t = 0; t < 4; ++t) {
      v8f a = zero_v8f();
      a = wmma16x16x32(qf[0], load_bfrag(&Ksm[t * 16 * 64], 64, lane, 0), a);
      a = wmma16x16x32(qf[1], load_bfrag(&Ksm[t * 16 * 64], 64, lane, 32), a);
      s[t] = a;
    }

    // --- scale, causal mask, online softmax (row = i+8g lives in one
    //     16-lane half => shfl_xor(1,2,4,8) is a full row reduction).
    const bool diag = (kb == qt);
    float p[4][8];
#pragma unroll
    for (int i = 0; i < 8; ++i) {
      const int qrow = qt * 64 + w * 16 + i + 8 * g;
      float mx = NEGBIG;
#pragma unroll
      for (int t = 0; t < 4; ++t) {
        float sv = s[t][i] * scale;
        int key = kb * 64 + t * 16 + nl;
        sv = (diag && key > qrow) ? NEGBIG : sv;
        p[t][i] = sv;
        mx = fmaxf(mx, sv);
      }
#pragma unroll
      for (int d = 1; d < 16; d <<= 1) mx = fmaxf(mx, __shfl_xor(mx, d, 32));
      float nm = fmaxf(mrow[i], mx);
      float fac = __expf(mrow[i] - nm);
      mrow[i] = nm;
      float rs = 0.f;
#pragma unroll
      for (int t = 0; t < 4; ++t) {
        float e = __expf(p[t][i] - nm);
        p[t][i] = e;
        rs += e;
      }
#pragma unroll
      for (int d = 1; d < 16; d <<= 1) rs += __shfl_xor(rs, d, 32);
      lrow[i] = lrow[i] * fac + rs;
#pragma unroll
      for (int dt = 0; dt < 4; ++dt) oacc[dt][i] *= fac;
    }

    // --- reshape P (C-layout) -> A-fragment layout via wave-private LDS.
#pragma unroll
    for (int t = 0; t < 4; ++t)
#pragma unroll
      for (int i = 0; i < 8; ++i)
        Psm[w][(i + 8 * g) * 64 + t * 16 + nl] = (_Float16)p[t][i];

    // --- O += P V.
#pragma unroll
    for (int ks2 = 0; ks2 < 2; ++ks2) {
      v16h af = load_afrag(Psm[w], 64, lane, ks2 * 32);
#pragma unroll
      for (int dt = 0; dt < 4; ++dt) {
        v16h bf = load_bfrag(&Vsm[dt * 16 * 64], 64, lane, ks2 * 32);
        oacc[dt] = wmma16x16x32(af, bf, oacc[dt]);
      }
    }
  }

  // --- normalize, write f16 [B*T][D_MODEL].
#pragma unroll
  for (int i = 0; i < 8; ++i) {
    float inv = 1.0f / lrow[i];
    int row = bT + qt * 64 + w * 16 + i + 8 * g;
#pragma unroll
    for (int dt = 0; dt < 4; ++dt)
      attn_out[(size_t)row * D_MODEL + h * HEAD_DIM + dt * 16 + nl] =
          (_Float16)(oacc[dt][i] * inv);
  }
}

// ---------------------------------------------------------------------------
extern "C" void kernel_launch(void* const* d_in, const int* in_sizes, int n_in,
                              void* d_out, int out_size, void* d_ws, size_t ws_size,
                              hipStream_t stream) {
  const float* x = (const float*)d_in[0];      // [2,2048,1024]
  const float* w_qkv = (const float*)d_in[1];  // [1024,3072]
  const float* b_qkv = (const float*)d_in[2];  // [3072]
  const float* w_out = (const float*)d_in[3];  // [1024,1024]
  const float* b_out = (const float*)d_in[4];  // [1024]
  float* out = (float*)d_out;                  // [2,2048,1024] f32

  const int BT = 2 * T_SEQ;  // 4096 token rows
  // Workspace layout (f16 elements), ~59 MB total:
  _Float16* x_h = (_Float16*)d_ws;                 // [4096][1024]
  _Float16* wqkvT = x_h + (size_t)BT * D_MODEL;    // [3072][1024]
  _Float16* woutT = wqkvT + (size_t)C3 * D_MODEL;  // [1024][1024]
  _Float16* qkv_h = woutT + (size_t)D_MODEL * D_MODEL;  // [4096][3072]
  _Float16* vT = qkv_h + (size_t)BT * C3;          // [2*16*64][2048]
  _Float16* attn_h = vT + (size_t)2 * N_HEADS * HEAD_DIM * T_SEQ;  // [4096][1024]

  // 0) f16 conversion passes.
  cvt_f32_to_f16<<<(BT * D_MODEL / 4 + 255) / 256, 256, 0, stream>>>(
      x, x_h, BT * D_MODEL / 4);
  transpose_cvt_f16<<<dim3(C3 / 64, D_MODEL / 64), 256, 0, stream>>>(
      w_qkv, wqkvT, D_MODEL, C3);
  transpose_cvt_f16<<<dim3(D_MODEL / 64, D_MODEL / 64), 256, 0, stream>>>(
      w_out, woutT, D_MODEL, D_MODEL);

  // 1) QKV projection (writes qkv_h f16 + transposed V into vT).
  gemm_f16<1><<<dim3(C3 / 64, BT / 128), 256, 0, stream>>>(
      x_h, wqkvT, b_qkv, (void*)qkv_h, vT, BT, C3, D_MODEL);

  // 2) Causal flash attention.
  attn_causal<<<dim3(T_SEQ / 64, N_HEADS, 2), 128, 0, stream>>>(
      qkv_h, vT, attn_h);

  // 3) Output projection (f32 out).
  gemm_f16<0><<<dim3(D_MODEL / 64, BT / 128), 256, 0, stream>>>(
      attn_h, woutT, b_out, out, nullptr, BT, D_MODEL, D_MODEL);
}
